// SpaceAttBlock_46566035423441
// MI455X (gfx1250) — matrix operand order repaired and checked by reading, not compiled, not run on hardware
//
#include <hip/hip_runtime.h>
#include <hip/hip_bf16.h>
#include <math.h>

#define B_   16
#define LK_  4096
#define D_   1024
#define H_   16
#define DK_  64
#define ROWS (B_ * LK_)      /* 65536 */
#define KDIM D_              /* 1024  */
#define KC_  (KDIM / 32)     /* 32 k-chunks of 32 */
#define NTW  4               /* 16x16 n-tiles per wave */

typedef __attribute__((ext_vector_type(16))) __bf16 v16bf;
typedef __attribute__((ext_vector_type(8)))  __bf16 v8bf;
typedef __attribute__((ext_vector_type(4)))  __bf16 v4bf;
typedef __attribute__((ext_vector_type(8)))  float  v8f;

// ---------------------------------------------------------------------------
// Prepack f32 weights into bf16 WMMA B-fragment layout.
// B fragment (32x16 bf16, wave32), this code's k order: lane = (k>=16)*16 + n ; element i: k = (lane>>4)*16 + i
// Packed address: ((ntile*KC + kc)*32 + lane)*16 + i  -> each lane's 16 elems contiguous.
// n < splitN comes from W0 (stride splitN); n >= splitN comes from W1 (stride 1024).
// ---------------------------------------------------------------------------
__global__ void prepack_kernel(const float* __restrict__ W0,
                               const float* __restrict__ W1,
                               int N, int splitN, __bf16* __restrict__ pk) {
    size_t gid = (size_t)blockIdx.x * blockDim.x + threadIdx.x;
    size_t total = (size_t)KDIM * N;
    if (gid >= total) return;
    int blk    = (int)(gid >> 9);
    int within = (int)(gid & 511);
    int lane   = within >> 4;
    int e      = within & 15;
    int kc = blk % KC_;
    int nt = blk / KC_;
    int k = kc * 32 + ((lane >> 4) << 4) + e;
    int n = (nt << 4) + (lane & 15);
    float v;
    if (n < splitN) v = W0[(size_t)k * splitN + n];
    else            v = W1[(size_t)k * 1024 + (n - splitN)];
    pk[gid] = (__bf16)v;
}

// ---------------------------------------------------------------------------
// Per-batch small projections: Q = query@Wq + bq ; qvp = query @ Wvp[1024:2048, :]
// ---------------------------------------------------------------------------
__global__ void qproj_kernel(const float* __restrict__ query,
                             const float* __restrict__ Wq, const float* __restrict__ bq,
                             const float* __restrict__ Wvp,
                             float* __restrict__ Qf, float* __restrict__ qvp) {
    const int b = blockIdx.x, t = threadIdx.x;
    __shared__ float q[D_];
    for (int j = t; j < D_; j += 256) q[j] = query[(size_t)b * D_ + j];
    __syncthreads();
    for (int jj = 0; jj < 4; ++jj) {
        int j = t + (jj << 8);
        float aq = bq[j], ap = 0.0f;
        for (int d = 0; d < D_; ++d) {
            float qd = q[d];
            aq += qd * Wq[(size_t)d * D_ + j];
            ap += qd * Wvp[(size_t)(D_ + d) * D_ + j];
        }
        Qf[(size_t)b * D_ + j]  = aq;
        qvp[(size_t)b * D_ + j] = ap;
    }
}

// ---------------------------------------------------------------------------
// WMMA GEMM: [ROWS x 1024] (f32 or bf16) @ packed bf16 [1024 x Ntot] -> bf16.
// 256 threads = 8 waves; workgroup tile = 16 rows x 512 cols (4 n-tiles/wave).
// Full 16x1024 A tile staged row-major in LDS (32 KB) with vectorized loads.
// A fragment per lane = two contiguous 8-elem runs: rows m=lane&15,
//   run0 at k = kk + g*16, run1 at k = kk + g*16 + 8   (g = lane>>4)
// -> two aligned ds_load_b128 per WMMA step, no barriers in the k-loop.
// Epilogue: col < splitN -> Out0 (+bias0) ; else Out1 (+qvp[batch]).
// ---------------------------------------------------------------------------
template <bool AF32>
__global__ void gemm_wmma_kernel(const void* __restrict__ Asrc,
                                 const __bf16* __restrict__ Bpk,
                                 __bf16* __restrict__ Out0,
                                 __bf16* __restrict__ Out1,
                                 const float* __restrict__ bias0,
                                 const float* __restrict__ qvp,
                                 int Ntot, int splitN) {
    const int tid  = threadIdx.x;
    const int lane = tid & 31;
    const int wave = tid >> 5;
    const int nBlocks = Ntot >> 9;              // 512 cols per workgroup
    const int rowTile = blockIdx.x / nBlocks;
    const int nblk    = blockIdx.x % nBlocks;
    const int row0   = rowTile << 4;
    const int n0     = (nblk << 9) + (wave << 6);  // wave covers 64 cols
    const int ntile0 = n0 >> 4;

    __shared__ __bf16 Asm[16 * KDIM];  // 32 KB row-major A tile

    // ---- stage A tile (vectorized: b128 global loads, b64 LDS stores) ----
    if constexpr (AF32) {
        const float* Af = (const float*)Asrc;
        for (int gq = tid; gq < (16 * KDIM) / 4; gq += 256) {
            int fl = gq << 2;
            int m  = fl >> 10;
            int kx = fl & (KDIM - 1);
            const float4 v = *(const float4*)(Af + (size_t)(row0 + m) * KDIM + kx);
            v4bf o = { (__bf16)v.x, (__bf16)v.y, (__bf16)v.z, (__bf16)v.w };
            *(v4bf*)&Asm[fl] = o;
        }
    } else {
        const __bf16* Ab = (const __bf16*)Asrc;
        for (int gq = tid; gq < (16 * KDIM) / 8; gq += 256) {
            int fl = gq << 3;
            int m  = fl >> 10;
            int kx = fl & (KDIM - 1);
            *(v8bf*)&Asm[fl] = *(const v8bf*)(Ab + (size_t)(row0 + m) * KDIM + kx);
        }
    }
    __syncthreads();

    // ---- k-loop: 2 ds_load_b128 (A) + 8 global b128 (B) + 4 WMMA per iter ----
    const int m = lane & 15;
    const int g = lane >> 4;
    const __bf16* arow = &Asm[m * KDIM + g * 16];

    v8f c[NTW] = {};
    union AFrag { v16bf v; v8bf h[2]; };

    for (int kc = 0; kc < KC_; ++kc) {
        const int kk = kc << 5;
        AFrag af;
        af.h[0] = *(const v8bf*)(arow + kk);
        af.h[1] = *(const v8bf*)(arow + kk + 8);
        const __bf16* bp0 =
            Bpk + ((((size_t)ntile0 * KC_ + kc) << 5) + lane) * 16;
        if (kc + 1 < KC_) __builtin_prefetch(bp0 + 512, 0, 0);  // next k-chunk
#pragma unroll
        for (int ti = 0; ti < NTW; ++ti) {
            const __bf16* bp = bp0 + ((size_t)ti * KC_ << 9);  // ti*KC*512 elems
            v16bf bm = *(const v16bf*)bp;
            c[ti] = __builtin_amdgcn_wmma_f32_16x16x32_bf16(
                false, af.v, false, bm, (short)0, c[ti], false, false);
        }
    }

    // ---- epilogue: C vgpr r -> M = r + 8*(lane>>4), N = tile_n0 + (lane&15) ----
    const int madd  = g << 3;
    const int batch = row0 >> 12;  // 4096 rows per batch, tiles never straddle
#pragma unroll
    for (int ti = 0; ti < NTW; ++ti) {
        const int ncol = n0 + (ti << 4) + (lane & 15);
#pragma unroll
        for (int r = 0; r < 8; ++r) {
            int row = row0 + r + madd;
            float val = c[ti][r];
            if (ncol < splitN) {
                if (bias0) val += bias0[ncol];
                Out0[(size_t)row * splitN + ncol] = (__bf16)val;
            } else {
                int n1 = ncol - splitN;
                val += qvp[(size_t)batch * D_ + n1];
                Out1[(size_t)row * (Ntot - splitN) + n1] = (__bf16)val;
            }
        }
    }
}

// ---------------------------------------------------------------------------
// Attention per (b,h): scores -> mask -> softplus threshold -> softmax -> attn@V
// 128 threads. If every score is masked to NEG, exp(0)=1 everywhere -> uniform
// softmax, exactly matching jax.nn.softmax of an all-equal row.
// ---------------------------------------------------------------------------
__global__ void attn_kernel(const float* __restrict__ Qf,
                            const __bf16* __restrict__ Kb,
                            const __bf16* __restrict__ Vb,
                            const int* __restrict__ mask,
                            const float* __restrict__ raw_thr,
                            float* __restrict__ xatt) {
    const int b = blockIdx.x >> 4;
    const int h = blockIdx.x & 15;
    const int t = threadIdx.x;  // 128 threads

    __shared__ float s[LK_];
    __shared__ float qh[DK_];
    __shared__ float red[128];

    if (t < DK_) qh[t] = Qf[(size_t)b * D_ + h * DK_ + t];
    __syncthreads();

    const float thr = log1pf(expf(raw_thr[h]));  // softplus
    const float NEG = -1.0e9f;

    float lmax = -3.4e38f;
    for (int l = t; l < LK_; l += 128) {
        const v8bf* kr = (const v8bf*)(Kb + ((size_t)(b * LK_ + l)) * D_ + h * DK_);
        float acc = 0.0f;
#pragma unroll
        for (int j = 0; j < 8; ++j) {
            v8bf kv = kr[j];
#pragma unroll
            for (int i = 0; i < 8; ++i) acc += qh[j * 8 + i] * (float)kv[i];
        }
        acc *= 0.125f;  // 1/sqrt(64)
        if (mask[(size_t)b * LK_ + l] == 0) acc = NEG;
        if (acc <= thr) acc = NEG;
        s[l] = acc;
        lmax = fmaxf(lmax, acc);
    }
    red[t] = lmax;
    __syncthreads();
    for (int off = 64; off > 0; off >>= 1) {
        if (t < off) red[t] = fmaxf(red[t], red[t + off]);
        __syncthreads();
    }
    const float m = red[0];
    __syncthreads();

    float lsum = 0.0f;
    for (int l = t; l < LK_; l += 128) {
        float e = expf(s[l] - m);
        s[l] = e;
        lsum += e;
    }
    red[t] = lsum;
    __syncthreads();
    for (int off = 64; off > 0; off >>= 1) {
        if (t < off) red[t] += red[t + off];
        __syncthreads();
    }
    const float Z = red[0];

    // weighted sum over V: thread = (half, d); each half covers 2048 keys
    const int d = t & 63;
    const int half = t >> 6;
    float acc = 0.0f;
    const int l0 = half * (LK_ / 2);
    for (int l = l0; l < l0 + (LK_ / 2); ++l)
        acc += s[l] * (float)Vb[((size_t)(b * LK_ + l)) * D_ + h * DK_ + d];
    __syncthreads();  // protect red[0] (Z) before reuse
    red[t] = acc;
    __syncthreads();
    if (t < 64) xatt[(size_t)b * D_ + h * DK_ + t] = (red[t] + red[t + 64]) / Z;
}

// ---------------------------------------------------------------------------
// Final: attn_out = x@Wo+bo ; LN(query+attn_out) ; FFN ; LN. One block per batch.
// ---------------------------------------------------------------------------
__device__ __forceinline__ float block_reduce_sum(float v, float* red) {
    const int t = threadIdx.x;
    red[t] = v;
    __syncthreads();
    for (int off = 128; off > 0; off >>= 1) {
        if (t < off) red[t] += red[t + off];
        __syncthreads();
    }
    float s = red[0];
    __syncthreads();
    return s;
}

__global__ void final_kernel(const float* __restrict__ query,
                             const float* __restrict__ xatt,
                             const float* __restrict__ Wo, const float* __restrict__ bo,
                             const float* __restrict__ W1, const float* __restrict__ b1,
                             const float* __restrict__ W2, const float* __restrict__ b2,
                             const float* __restrict__ g1, const float* __restrict__ be1,
                             const float* __restrict__ g2, const float* __restrict__ be2,
                             float* __restrict__ out) {
    const int b = blockIdx.x, t = threadIdx.x;  // 256 threads
    __shared__ float lx[D_];
    __shared__ float h1[D_];
    __shared__ float hid[2 * D_];
    __shared__ float red[256];

    for (int j = t; j < D_; j += 256) lx[j] = xatt[(size_t)b * D_ + j];
    __syncthreads();

    float r[4];
#pragma unroll
    for (int jj = 0; jj < 4; ++jj) {
        int j = t + (jj << 8);
        float acc = bo[j];
        for (int d = 0; d < D_; ++d) acc += lx[d] * Wo[(size_t)d * D_ + j];
        r[jj] = query[(size_t)b * D_ + j] + acc;
    }
    float mean = block_reduce_sum(r[0] + r[1] + r[2] + r[3], red) * (1.0f / D_);
    float vs = 0.0f;
#pragma unroll
    for (int jj = 0; jj < 4; ++jj) { float d = r[jj] - mean; vs += d * d; }
    float rstd = rsqrtf(block_reduce_sum(vs, red) * (1.0f / D_) + 1e-5f);
#pragma unroll
    for (int jj = 0; jj < 4; ++jj) {
        int j = t + (jj << 8);
        h1[j] = (r[jj] - mean) * rstd * g1[j] + be1[j];
    }
    __syncthreads();

    for (int k = t; k < 2 * D_; k += 256) {
        float acc = b1[k];
        for (int j = 0; j < D_; ++j) acc += h1[j] * W1[(size_t)j * (2 * D_) + k];
        hid[k] = fmaxf(acc, 0.0f);
    }
    __syncthreads();

    float r2[4];
#pragma unroll
    for (int jj = 0; jj < 4; ++jj) {
        int j = t + (jj << 8);
        float acc = b2[j];
        for (int k = 0; k < 2 * D_; ++k) acc += hid[k] * W2[(size_t)k * D_ + j];
        r2[jj] = h1[j] + acc;
    }
    float mean2 = block_reduce_sum(r2[0] + r2[1] + r2[2] + r2[3], red) * (1.0f / D_);
    vs = 0.0f;
#pragma unroll
    for (int jj = 0; jj < 4; ++jj) { float d = r2[jj] - mean2; vs += d * d; }
    float rstd2 = rsqrtf(block_reduce_sum(vs, red) * (1.0f / D_) + 1e-5f);
#pragma unroll
    for (int jj = 0; jj < 4; ++jj) {
        int j = t + (jj << 8);
        out[(size_t)b * D_ + j] = (r2[jj] - mean2) * rstd2 * g2[j] + be2[j];
    }
}

// ---------------------------------------------------------------------------
extern "C" void kernel_launch(void* const* d_in, const int* in_sizes, int n_in,
                              void* d_out, int out_size, void* d_ws, size_t ws_size,
                              hipStream_t stream) {
    const float* query   = (const float*)d_in[0];
    const float* key     = (const float*)d_in[1];
    const int*   mask    = (const int*)d_in[2];
    const float* Wq      = (const float*)d_in[3];
    const float* bq      = (const float*)d_in[4];
    const float* Wk      = (const float*)d_in[5];
    const float* bk      = (const float*)d_in[6];
    const float* Wv      = (const float*)d_in[7];
    const float* bv      = (const float*)d_in[8];
    const float* Wo      = (const float*)d_in[9];
    const float* bo      = (const float*)d_in[10];
    const float* raw_thr = (const float*)d_in[11];
    const float* Wvp     = (const float*)d_in[12];
    const float* W1      = (const float*)d_in[13];
    const float* b1      = (const float*)d_in[14];
    const float* W2      = (const float*)d_in[15];
    const float* b2      = (const float*)d_in[16];
    const float* g1      = (const float*)d_in[17];
    const float* be1     = (const float*)d_in[18];
    const float* g2      = (const float*)d_in[19];
    const float* be2     = (const float*)d_in[20];

    char* ws = (char*)d_ws;
    size_t off = 0;
    __bf16* WcatPk = (__bf16*)(ws + off); off += (size_t)1024 * 2048 * 2;  // packed [Wk | Wvp_k]
    __bf16* WvPk   = (__bf16*)(ws + off); off += (size_t)1024 * 1024 * 2;  // packed Wv
    float*  Qf     = (float*)(ws + off);  off += (size_t)B_ * D_ * 4;
    float*  qvp    = (float*)(ws + off);  off += (size_t)B_ * D_ * 4;
    float*  xatt   = (float*)(ws + off);  off += (size_t)B_ * D_ * 4;
    __bf16* Kb     = (__bf16*)(ws + off); off += (size_t)ROWS * D_ * 2;    // K proj (bf16)
    __bf16* Vt     = (__bf16*)(ws + off); off += (size_t)ROWS * D_ * 2;    // v = key@Wvp_k + qvp
    __bf16* Vb     = (__bf16*)(ws + off); off += (size_t)ROWS * D_ * 2;    // V = v@Wv + bv

    prepack_kernel<<<(1024 * 2048) / 256, 256, 0, stream>>>(Wk, Wvp, 2048, 1024, WcatPk);
    prepack_kernel<<<(1024 * 1024) / 256, 256, 0, stream>>>(Wv, nullptr, 1024, 1024, WvPk);
    qproj_kernel<<<B_, 256, 0, stream>>>(query, Wq, bq, Wvp, Qf, qvp);

    // K = key@Wk + bk ; Vt = key@Wvp_k + qvp  (fused: Ntot=2048, split at 1024)
    gemm_wmma_kernel<true><<<(ROWS / 16) * (2048 / 512), 256, 0, stream>>>(
        key, WcatPk, Kb, Vt, bk, qvp, 2048, 1024);
    // V = Vt@Wv + bv
    gemm_wmma_kernel<false><<<(ROWS / 16) * (1024 / 512), 256, 0, stream>>>(
        Vt, WvPk, Vb, nullptr, bv, nullptr, 1024, 1024);

    attn_kernel<<<B_ * H_, 128, 0, stream>>>(Qf, Kb, Vb, mask, raw_thr, xatt);
    final_kernel<<<B_, 256, 0, stream>>>(query, xatt, Wo, bo, W1, b1, W2, b2,
                                         g1, be1, g2, be2, (float*)d_out);
}
